// BlockSparseAttention_64201171141507
// MI455X (gfx1250) — compile-verified, matrix operand order checked by reading, not run on hardware
//
#include <hip/hip_runtime.h>

// ---------------------------------------------------------------------------
// Block-sparse attention for MI455X (gfx1250), wave32 + WMMA bf16 path.
//   1) cvt x -> bf16; cvt+transpose Wq/Wk/Wv/Wo -> bf16 [N][K]
//   2) WMMA GEMM (async-to-LDS double-buffered, pipelined B-fragments):
//        Q=(x@Wq+bq) -> (b,h,n,dh) bf16
//        K=(x@Wk+bk) -> (b,h,n,dh) bf16
//        V=(x@Wv+bv) -> (b,h,dh,n) bf16
//   3) flash-style block attention, K/V tiles staged in LDS via async copy
//   4) WMMA GEMM: out = O@Wo + bo -> f32
// ---------------------------------------------------------------------------

#define D_MODEL 1024
#define HEADS   16
#define DH      64
#define BZ      64
#define SEQ     4096
#define BATCH   4
#define NBLK    64            // SEQ / BZ
#define MROWS   (BATCH * SEQ) // 16384

typedef __bf16 bf16;
typedef __bf16        v16bf  __attribute__((ext_vector_type(16)));
typedef float         v8f    __attribute__((ext_vector_type(8)));
typedef unsigned int  u32x4  __attribute__((ext_vector_type(4)));
typedef int           i32x4g __attribute__((__vector_size__(16))); // builtin's pointee type

#define GLOBAL_AS __attribute__((address_space(1)))
#define LDS_AS    __attribute__((address_space(3)))

#if defined(__AMDGCN__) && __has_builtin(__builtin_amdgcn_global_load_async_to_lds_b128)
#define HAVE_ASYNC_LDS 1
#else
#define HAVE_ASYNC_LDS 0
#endif

union Frag { v16bf v; u32x4 u[2]; };

__device__ __forceinline__ v8f v8f_zero() {
  v8f z;
#pragma unroll
  for (int i = 0; i < 8; ++i) z[i] = 0.0f;
  return z;
}

// 16-byte global -> LDS copy. Async (no VGPR roundtrip, ASYNCcnt) when the
// gfx1250 builtin is available; plain load/store otherwise.
__device__ __forceinline__ void copy16(const bf16* g, bf16* l) {
#if HAVE_ASYNC_LDS
  __builtin_amdgcn_global_load_async_to_lds_b128(
      (GLOBAL_AS i32x4g*)g,
      (LDS_AS i32x4g*)l, 0, 0);
#else
  *reinterpret_cast<u32x4*>(l) = *reinterpret_cast<const u32x4*>(g);
#endif
}

__device__ __forceinline__ void wait_async_le4() {
#if HAVE_ASYNC_LDS
#if __has_builtin(__builtin_amdgcn_s_wait_asynccnt)
  __builtin_amdgcn_s_wait_asynccnt(4);
#else
  asm volatile("s_wait_asynccnt 4" ::: "memory");
#endif
#endif
}

__device__ __forceinline__ void wait_async_0() {
#if HAVE_ASYNC_LDS
#if __has_builtin(__builtin_amdgcn_s_wait_asynccnt)
  __builtin_amdgcn_s_wait_asynccnt(0);
#else
  asm volatile("s_wait_asynccnt 0" ::: "memory");
#endif
#endif
}

// Load one 16x32 (A) or [N=16]x[K=32] (B, N-major memory) bf16 fragment.
// ISA layout: lanes 0-15 hold row (lane), K = {0..7, 16..23};
//             lanes 16-31 hold row (lane-16), K = {8..15, 24..31}.
__device__ __forceinline__ Frag load_frag16(const bf16* base, int stride) {
  Frag f;
  const int lane = threadIdx.x & 31;
  const bf16* p = base + (lane & 15) * stride + ((lane >> 4) << 3);
  f.u[0] = *reinterpret_cast<const u32x4*>(p);
  f.u[1] = *reinterpret_cast<const u32x4*>(p + 16);
  return f;
}

__device__ __forceinline__ v8f wmma_bf16(v16bf a, v16bf b, v8f c) {
  return __builtin_amdgcn_wmma_f32_16x16x32_bf16(false, a, false, b,
                                                 (short)0, c, false, false);
}

// ---------------------------------------------------------------------------
// Conversion kernels
// ---------------------------------------------------------------------------
__global__ __launch_bounds__(256) void cvt_bf16_kernel(
    const float* __restrict__ in, bf16* __restrict__ out, int n) {
  int i = blockIdx.x * 256 + threadIdx.x;
  if (i < n) out[i] = (bf16)in[i];
}

// out[n * rows + k] = in[k * cols + n]
__global__ __launch_bounds__(256) void transpose_bf16_kernel(
    const float* __restrict__ in, bf16* __restrict__ out, int rows, int cols) {
  int i = blockIdx.x * 256 + threadIdx.x;
  if (i < rows * cols) {
    int k = i / cols;
    int n = i - k * cols;
    out[n * rows + k] = (bf16)in[i];
  }
}

// ---------------------------------------------------------------------------
// bf16 WMMA GEMM:  C[M,N] = A[M,K] * Bt[N,K]^T + bias
// WG tile 128x128, 8 waves, K step 32, double-buffered async-to-LDS staging,
// one-stage software pipeline on B fragments.
// mode 0: bf16 -> (b,h,n,dh);  mode 1: bf16 -> (b,h,dh,n);  mode 2: f32 [M,N]
// ---------------------------------------------------------------------------
__global__ __launch_bounds__(256) void gemm_bf16_kernel(
    const bf16* __restrict__ A, const bf16* __restrict__ Bt,
    const float* __restrict__ bias, void* __restrict__ outp,
    int M, int N, int K, int mode) {
  const int n0   = blockIdx.x * 128;
  const int m0   = blockIdx.y * 128;
  const int tid  = threadIdx.x;
  const int wave = tid >> 5;
  const int lane = tid & 31;
  const int half = lane >> 4;

  __shared__ bf16 As[2][128 * 32]; // [m][k], 8 KB per buffer
  __shared__ bf16 Bs[2][128 * 32]; // [n][k], 8 KB per buffer

  v8f acc[8];
#pragma unroll
  for (int i = 0; i < 8; ++i) acc[i] = v8f_zero();

  // Issue one 128x32 A-tile + 128x32 B-tile copy: 4 x 16B per thread.
  auto issue_tile = [&](int buf, int kk) {
#pragma unroll
    for (int i = 0; i < 2; ++i) {
      int c   = tid + i * 256;  // 0..511
      int row = c >> 2;         // 4 chunks of 16B per 32-elem row
      int cc  = (c & 3) * 8;
      copy16(&A[(size_t)(m0 + row) * K + kk + cc],  &As[buf][row * 32 + cc]);
      copy16(&Bt[(size_t)(n0 + row) * K + kk + cc], &Bs[buf][row * 32 + cc]);
    }
  };

  const int nk = K >> 5;
  issue_tile(0, 0);
  for (int t = 0; t < nk; ++t) {
    const int cur = t & 1;
    if (t + 1 < nk) {
      issue_tile(cur ^ 1, (t + 1) << 5); // prefetch next tile
      wait_async_le4();                  // in-order: current tile has landed
    } else {
      wait_async_0();
    }
    __syncthreads();

    Frag a    = load_frag16(&As[cur][wave * 16 * 32], 32);
    Frag bcur = load_frag16(&Bs[cur][0], 32);
#pragma unroll
    for (int nt = 0; nt < 8; ++nt) {
      Frag bnext = (nt < 7) ? load_frag16(&Bs[cur][(nt + 1) * 16 * 32], 32)
                            : bcur;
      acc[nt] = wmma_bf16(a.v, bcur.v, acc[nt]);
      bcur = bnext;
    }
    __syncthreads(); // buffer `cur` free for reuse two iterations later
  }

  // Epilogue: C layout — lane holds (row = r + 8*half, col = lane&15).
#pragma unroll
  for (int nt = 0; nt < 8; ++nt) {
    int col  = n0 + nt * 16 + (lane & 15);
    float bv = bias[col];
#pragma unroll
    for (int r = 0; r < 8; ++r) {
      int row   = m0 + wave * 16 + r + half * 8;
      float val = acc[nt][r] + bv;
      if (mode == 2) {
        reinterpret_cast<float*>(outp)[(size_t)row * N + col] = val;
      } else {
        int bb = row >> 12, nn = row & 4095; // b, n  (SEQ = 4096)
        int hh = col >> 6,  dh = col & 63;   // h, dh (DH  = 64)
        bf16* o = reinterpret_cast<bf16*>(outp);
        size_t idx = (mode == 0)
            ? ((((size_t)bb * HEADS + hh) * SEQ + nn) * DH + dh)
            : ((((size_t)bb * HEADS + hh) * DH + dh) * SEQ + nn);
        o[idx] = (bf16)val;
      }
    }
  }
}

// ---------------------------------------------------------------------------
// Flash-style block attention.
// grid = (qblock, bh), 128 threads (4 waves); wave w owns 16 query rows.
// K/V 64x64 tiles staged in LDS (async copy), shared by all 4 waves.
// ---------------------------------------------------------------------------
__global__ __launch_bounds__(128) void attn_kernel(
    const bf16* __restrict__ Q,   // (b,h,n,dh)
    const bf16* __restrict__ Kn,  // (b,h,n,dh)
    const bf16* __restrict__ Vt,  // (b,h,dh,n)
    bf16* __restrict__ O) {       // (b,n,h*DH+dh)
  const int qb   = blockIdx.x;
  const int bh   = blockIdx.y;
  const int b    = bh >> 4;
  const int h    = bh & 15;
  const int wave = threadIdx.x >> 5;
  const int lane = threadIdx.x & 31;
  const int half = lane >> 4;

  __shared__ bf16 Ks[64 * 64];      // [key][dh]   8 KB
  __shared__ bf16 Vs[64 * 64];      // [dh][key]   8 KB
  __shared__ bf16 pbuf[4][16 * 64]; // per-wave P staging, 8 KB

  const size_t hdBase = (size_t)bh * SEQ * DH;

  const bf16* qrow = Q + hdBase + (size_t)(qb * BZ + wave * 16) * DH;
  Frag qf0 = load_frag16(qrow,      DH);
  Frag qf1 = load_frag16(qrow + 32, DH);

  v8f o[4];
#pragma unroll
  for (int i = 0; i < 4; ++i) o[i] = v8f_zero();

  float rm[8], rs[8];
#pragma unroll
  for (int r = 0; r < 8; ++r) { rm[r] = -1e30f; rs[r] = 0.0f; }

  const int nkb = (qb == 0) ? NBLK : ((qb == NBLK - 1) ? 2 : 3);
  const int kb0 = (qb == 0) ? 0 : qb - 1;
  const float scale = 0.125f; // 1/sqrt(64)

  for (int it = 0; it < nkb; ++it) {
    const int kblk = kb0 + it;

    // ---- stage K (64x64, [key][dh]) and V (64x64, [dh][key]) into LDS ----
    const bf16* kg = Kn + hdBase + (size_t)kblk * BZ * DH;
    const bf16* vg = Vt + (size_t)bh * DH * SEQ + kblk * BZ;
#pragma unroll
    for (int i = 0; i < 4; ++i) {
      int c   = threadIdx.x + i * 128; // 0..511
      int row = c >> 3;                // 8 chunks of 16B per 64-elem row
      int cc  = (c & 7) * 8;
      copy16(&kg[row * DH + cc],          &Ks[row * 64 + cc]);
      copy16(&vg[(size_t)row * SEQ + cc], &Vs[row * 64 + cc]);
    }
    wait_async_0();
    __syncthreads();

    // ---- S = Q * K^T : this wave's 16x64 strip = 4 tiles (pipelined) ----
    v8f s[4];
    Frag kf0 = load_frag16(&Ks[0],  64);
    Frag kf1 = load_frag16(&Ks[32], 64);
#pragma unroll
    for (int nt = 0; nt < 4; ++nt) {
      Frag kn0 = (nt < 3) ? load_frag16(&Ks[(nt + 1) * 16 * 64],      64) : kf0;
      Frag kn1 = (nt < 3) ? load_frag16(&Ks[(nt + 1) * 16 * 64 + 32], 64) : kf1;
      v8f t = v8f_zero();
      t = wmma_bf16(qf0.v, kf0.v, t);
      t = wmma_bf16(qf1.v, kf1.v, t);
#pragma unroll
      for (int r = 0; r < 8; ++r) s[nt][r] = t[r] * scale;
      kf0 = kn0;
      kf1 = kn1;
    }

    // ---- online softmax (row = r + 8*half, 16 cols across 16 lanes) ----
    float nm[8];
#pragma unroll
    for (int r = 0; r < 8; ++r) {
      nm[r] = rm[r];
#pragma unroll
      for (int nt = 0; nt < 4; ++nt) nm[r] = fmaxf(nm[r], s[nt][r]);
      nm[r] = fmaxf(nm[r], __shfl_xor(nm[r], 1, 32));
      nm[r] = fmaxf(nm[r], __shfl_xor(nm[r], 2, 32));
      nm[r] = fmaxf(nm[r], __shfl_xor(nm[r], 4, 32));
      nm[r] = fmaxf(nm[r], __shfl_xor(nm[r], 8, 32));
    }
    float al[8], ps[8];
#pragma unroll
    for (int r = 0; r < 8; ++r) {
      al[r] = __expf(rm[r] - nm[r]);
      rm[r] = nm[r];
      ps[r] = 0.0f;
    }
#pragma unroll
    for (int nt = 0; nt < 4; ++nt)
#pragma unroll
      for (int r = 0; r < 8; ++r) {
        float p = __expf(s[nt][r] - rm[r]);
        s[nt][r] = p;
        ps[r] += p;
      }
#pragma unroll
    for (int r = 0; r < 8; ++r) {
      ps[r] += __shfl_xor(ps[r], 1, 32);
      ps[r] += __shfl_xor(ps[r], 2, 32);
      ps[r] += __shfl_xor(ps[r], 4, 32);
      ps[r] += __shfl_xor(ps[r], 8, 32);
      rs[r] = rs[r] * al[r] + ps[r];
    }
#pragma unroll
    for (int nt = 0; nt < 4; ++nt)
#pragma unroll
      for (int r = 0; r < 8; ++r) o[nt][r] *= al[r];

    // ---- P: C-layout -> bf16 A-layout via per-wave LDS ----
#pragma unroll
    for (int nt = 0; nt < 4; ++nt)
#pragma unroll
      for (int r = 0; r < 8; ++r)
        pbuf[wave][(r + half * 8) * 64 + nt * 16 + (lane & 15)] =
            (bf16)s[nt][r];
    __syncthreads();

    Frag pf0 = load_frag16(&pbuf[wave][0],  64);
    Frag pf1 = load_frag16(&pbuf[wave][32], 64);

    // ---- O += P * V (V fragments from LDS, pipelined) ----
    Frag vf0 = load_frag16(&Vs[0],  64);
    Frag vf1 = load_frag16(&Vs[32], 64);
#pragma unroll
    for (int nt = 0; nt < 4; ++nt) {
      Frag vn0 = (nt < 3) ? load_frag16(&Vs[(nt + 1) * 16 * 64],      64) : vf0;
      Frag vn1 = (nt < 3) ? load_frag16(&Vs[(nt + 1) * 16 * 64 + 32], 64) : vf1;
      o[nt] = wmma_bf16(pf0.v, vf0.v, o[nt]);
      o[nt] = wmma_bf16(pf1.v, vf1.v, o[nt]);
      vf0 = vn0;
      vf1 = vn1;
    }
    __syncthreads(); // Ks/Vs free for next key block
  }

  // ---- normalize and write O in (b, n, h*DH+dh) bf16 ----
  float inv[8];
#pragma unroll
  for (int r = 0; r < 8; ++r) inv[r] = 1.0f / rs[r];
#pragma unroll
  for (int nt = 0; nt < 4; ++nt)
#pragma unroll
    for (int r = 0; r < 8; ++r) {
      int n = qb * BZ + wave * 16 + r + half * 8;
      int d = h * DH + nt * 16 + (lane & 15);
      O[((size_t)b * SEQ + n) * D_MODEL + d] = (bf16)(o[nt][r] * inv[r]);
    }
}

// ---------------------------------------------------------------------------
// Host-side launcher
// ---------------------------------------------------------------------------
extern "C" void kernel_launch(void* const* d_in, const int* in_sizes, int n_in,
                              void* d_out, int out_size, void* d_ws,
                              size_t ws_size, hipStream_t stream) {
  (void)in_sizes; (void)n_in; (void)out_size; (void)ws_size;

  const float* x  = (const float*)d_in[0];
  const float* Wq = (const float*)d_in[1];
  const float* bq = (const float*)d_in[2];
  const float* Wk = (const float*)d_in[3];
  const float* bk = (const float*)d_in[4];
  const float* Wv = (const float*)d_in[5];
  const float* bv = (const float*)d_in[6];
  const float* Wo = (const float*)d_in[7];
  const float* bo = (const float*)d_in[8];

  const size_t NX = (size_t)MROWS * D_MODEL;   // 16,777,216
  const size_t NW = (size_t)D_MODEL * D_MODEL; //  1,048,576

  char* ws = (char*)d_ws;
  size_t off = 0;
  bf16* xb  = (bf16*)(ws + off); off += NX * 2;
  bf16* Wqt = (bf16*)(ws + off); off += NW * 2;
  bf16* Wkt = (bf16*)(ws + off); off += NW * 2;
  bf16* Wvt = (bf16*)(ws + off); off += NW * 2;
  bf16* Wot = (bf16*)(ws + off); off += NW * 2;
  bf16* Qb  = (bf16*)(ws + off); off += NX * 2; // (b,h,n,dh)
  bf16* Kb  = (bf16*)(ws + off); off += NX * 2; // (b,h,n,dh)
  bf16* Vtb = (bf16*)(ws + off); off += NX * 2; // (b,h,dh,n)
  bf16* Ob  = (bf16*)(ws + off); off += NX * 2; // (b,n,d)

  cvt_bf16_kernel<<<(int)(NX / 256), 256, 0, stream>>>(x, xb, (int)NX);
  transpose_bf16_kernel<<<(int)(NW / 256), 256, 0, stream>>>(Wq, Wqt, D_MODEL, D_MODEL);
  transpose_bf16_kernel<<<(int)(NW / 256), 256, 0, stream>>>(Wk, Wkt, D_MODEL, D_MODEL);
  transpose_bf16_kernel<<<(int)(NW / 256), 256, 0, stream>>>(Wv, Wvt, D_MODEL, D_MODEL);
  transpose_bf16_kernel<<<(int)(NW / 256), 256, 0, stream>>>(Wo, Wot, D_MODEL, D_MODEL);

  dim3 gg(D_MODEL / 128, MROWS / 128); // (8, 128)
  gemm_bf16_kernel<<<gg, 256, 0, stream>>>(xb, Wqt, bq, Qb,  MROWS, D_MODEL, D_MODEL, 0);
  gemm_bf16_kernel<<<gg, 256, 0, stream>>>(xb, Wkt, bk, Kb,  MROWS, D_MODEL, D_MODEL, 0);
  gemm_bf16_kernel<<<gg, 256, 0, stream>>>(xb, Wvt, bv, Vtb, MROWS, D_MODEL, D_MODEL, 1);

  attn_kernel<<<dim3(NBLK, BATCH * HEADS), 128, 0, stream>>>(Qb, Kb, Vtb, Ob);

  gemm_bf16_kernel<<<gg, 256, 0, stream>>>(Ob, Wot, bo, d_out, MROWS, D_MODEL, D_MODEL, 2);
}
